// AttentionLayer4AUGRU_23682449670332
// MI455X (gfx1250) — matrix operand order repaired and checked by reading, not compiled, not run on hardware
//
#include <hip/hip_runtime.h>
#include <hip/hip_bf16.h>

typedef __attribute__((ext_vector_type(16))) _Float16 v16h;
typedef __attribute__((ext_vector_type(8)))  _Float16 v8h;
typedef __attribute__((ext_vector_type(8)))  float    v8f;

#define NEG_BIG (-4294967295.0f)

// Problem constants (fixed by the reference)
#define TT     200
#define DD     32
#define HH1    80
#define HH2    40
#define MPAD   208      // 13 tiles of 16
#define KA     128      // info width
#define PA     136      // A row pitch (halfs)
#define PH     104      // H row pitch (halfs)
#define PW2    104      // W2T row pitch (halfs)
#define N2PAD  48       // H2 padded to 3 tiles

// LDS layout (bytes)
#define OFF_A    0
#define SZ_A     (MPAD * PA * 2)            // 56576
#define OFF_H    (OFF_A + SZ_A)
#define SZ_H     (MPAD * PH * 2)            // 43264
#define OFF_W1   (OFF_H + SZ_H)
#define SZ_W1    (HH1 * PA * 2)             // 21760
#define OFF_W2   (OFF_W1 + SZ_W1)
#define SZ_W2    (N2PAD * PW2 * 2)          // 9984
#define OFF_WF   (OFF_W2 + SZ_W2)
#define SZ_WF    (N2PAD * 4)                // 192
#define OFF_QP   (OFF_WF + SZ_WF)
#define SZ_QP    (DD * 4)                   // 128
#define OFF_LG   (OFF_QP + SZ_QP)
#define SZ_LG    (MPAD * 4)                 // 832
#define SMEM_BYTES (OFF_LG + SZ_LG)         // 132736

// Fast sigmoid: v_exp_f32 + v_rcp_f32 (no IEEE divide sequence).
// Feeds f16 stores / softmax, so ~1 ulp rcp is ample.
__device__ __forceinline__ float sigm(float x) {
    return __builtin_amdgcn_rcpf(1.0f + __expf(-x));
}

// A fragment: lane holds row (lane&15); halfs 0..7 at K=kbase+8*(lane>>4),
// halfs 8..15 at K=kbase+16+8*(lane>>4). (ISA 7.12.2, 16-bit A 16x32)
__device__ __forceinline__ v16h load_a_frag(const _Float16* base, int pitch,
                                            int row, int kbase, int lane) {
    const _Float16* p = base + row * pitch + kbase + ((lane >> 4) << 3);
    v8h lo = *(const v8h*)p;
    v8h hi = *(const v8h*)(p + 16);
    v16h a;
#pragma unroll
    for (int i = 0; i < 8; ++i) { a[i] = lo[i]; a[i + 8] = hi[i]; }
    return a;
}

// B fragment from an N-major (transposed) weight: lane holds col (lane&15);
// halfs 0..7 at K=kbase+8*(lane>>4), halfs 8..15 at K=kbase+16+8*(lane>>4).
__device__ __forceinline__ v16h load_b_frag(const _Float16* base, int pitch,
                                            int col, int kbase, int lane) {
    const _Float16* p = base + col * pitch + kbase + ((lane >> 4) << 3);
    v8h lo = *(const v8h*)p;
    v8h hi = *(const v8h*)(p + 16);
    v16h b;
#pragma unroll
    for (int i = 0; i < 8; ++i) { b[i] = lo[i]; b[i + 8] = hi[i]; }
    return b;
}

extern "C" __global__ void __launch_bounds__(256)
din_attention_kernel(const float* __restrict__ qg,
                     const float* __restrict__ kg,
                     const int*   __restrict__ maskg,
                     const float* __restrict__ Wqg,
                     const float* __restrict__ bqg,
                     const float* __restrict__ alphag,
                     const float* __restrict__ W1g,
                     const float* __restrict__ b1g,
                     const float* __restrict__ W2g,
                     const float* __restrict__ b2g,
                     const float* __restrict__ Wfg,
                     const float* __restrict__ bfg,
                     float* __restrict__ outg) {
    extern __shared__ char smem[];
    _Float16* As   = (_Float16*)(smem + OFF_A);
    _Float16* Hs   = (_Float16*)(smem + OFF_H);
    _Float16* W1Ts = (_Float16*)(smem + OFF_W1);
    _Float16* W2Ts = (_Float16*)(smem + OFF_W2);
    float*    WfS  = (float*)(smem + OFF_WF);
    float*    qpS  = (float*)(smem + OFF_QP);
    float*    lgS  = (float*)(smem + OFF_LG);

    const int tid  = threadIdx.x;
    const int lane = tid & 31;
    const int wave = tid >> 5;
    const int b    = blockIdx.x;

    const float* kb = kg + (size_t)b * TT * DD;

    // Warm L2/L0 for this batch's k slab (global_prefetch_b8), 128B per line.
    if (tid < 200) __builtin_prefetch(kb + tid * 32, 0, 3);

    // ---------------- Phase 0: weights -> LDS (f16, transposed), qp ----------
    if (tid < 32) {
        // qp[j] = prelu(dot(q[b,:], Wq[:,j]) + bq[j])
        float acc = bqg[tid];
#pragma unroll
        for (int i = 0; i < DD; ++i) acc += qg[(size_t)b * DD + i] * Wqg[i * DD + tid];
        qpS[tid] = (acc >= 0.0f) ? acc : alphag[tid] * acc;
    }
    // W1T[n][k], pitch PA, zero K-pad (each element written exactly once)
    for (int idx = tid; idx < HH1 * PA; idx += 256) {
        int n = idx / PA, kk = idx - n * PA;
        _Float16 v = (kk < KA) ? (_Float16)W1g[kk * HH1 + n] : (_Float16)0.0f;
        W1Ts[n * PA + kk] = v;
    }
    // W2T[n][k], pitch PW2, zero pads (n>=40 or k>=80)
    for (int idx = tid; idx < N2PAD * PW2; idx += 256) {
        int n = idx / PW2, kk = idx - n * PW2;
        _Float16 v = (n < HH2 && kk < HH1) ? (_Float16)W2g[kk * HH2 + n] : (_Float16)0.0f;
        W2Ts[n * PW2 + kk] = v;
    }
    // Wf padded, logits init with bf, H pad cols 80..103 zero
    for (int idx = tid; idx < N2PAD; idx += 256) WfS[idx] = (idx < HH2) ? Wfg[idx] : 0.0f;
    for (int idx = tid; idx < MPAD; idx += 256) lgS[idx] = bfg[0];
    for (int idx = tid; idx < MPAD * 24; idx += 256) {
        int t = idx / 24, c = HH1 + (idx - t * 24);
        Hs[t * PH + c] = (_Float16)0.0f;
    }
    __syncthreads();

    // ---------------- Phase 1: build info matrix A (f16) ---------------------
    const float4* k4 = (const float4*)kb;
    for (int idx = tid; idx < MPAD * 8; idx += 256) {
        int t = idx >> 3, c4 = idx & 7;
        float4 kv = make_float4(0.f, 0.f, 0.f, 0.f);
        if (t < TT) kv = k4[t * 8 + c4];
        _Float16* Ar = As + t * PA;
        float kvs[4] = {kv.x, kv.y, kv.z, kv.w};
#pragma unroll
        for (int j = 0; j < 4; ++j) {
            int d = c4 * 4 + j;
            float qv = (t < TT) ? qpS[d] : 0.0f;
            float kj = kvs[j];
            Ar[d]          = (_Float16)qv;
            Ar[32 + d]     = (_Float16)kj;
            Ar[64 + d]     = (_Float16)(qv - kj);
            Ar[96 + d]     = (_Float16)(qv * kj);
        }
    }
    __syncthreads();

    // ---------------- GEMM1: [208x128] @ [128x80] -> sigmoid -> H ------------
    for (int tile = wave; tile < 13 * 5; tile += 8) {
        int mt = tile % 13, nt = tile / 13;
        int row = mt * 16 + (lane & 15);
        int col = nt * 16 + (lane & 15);
        v8f c = {};
#pragma unroll
        for (int ks = 0; ks < 4; ++ks) {
            v16h a  = load_a_frag(As,   PA, row, ks * 32, lane);
            v16h bb = load_b_frag(W1Ts, PA, col, ks * 32, lane);
            c = __builtin_amdgcn_wmma_f32_16x16x32_f16(false, a, false, bb,
                                                       (short)0, c, false, false);
        }
        float b1v = b1g[col];                 // col < 80 always
        int mbase = mt * 16 + ((lane >> 4) << 3);
#pragma unroll
        for (int r = 0; r < 8; ++r)
            Hs[(mbase + r) * PH + col] = (_Float16)sigm(c[r] + b1v);
    }
    __syncthreads();

    // ------- GEMM2: [208x80] @ [80x40] -> sigmoid -> dot Wf -> logits --------
    for (int tile = wave; tile < 13 * 3; tile += 8) {
        int mt = tile % 13, nt = tile / 13;
        int row = mt * 16 + (lane & 15);
        int col = nt * 16 + (lane & 15);     // 0..47 (pads contribute 0)
        v8f c = {};
#pragma unroll
        for (int ks = 0; ks < 3; ++ks) {
            v16h a  = load_a_frag(Hs,   PH,  row, ks * 32, lane);
            v16h bb = load_b_frag(W2Ts, PW2, col, ks * 32, lane);
            c = __builtin_amdgcn_wmma_f32_16x16x32_f16(false, a, false, bb,
                                                       (short)0, c, false, false);
        }
        float b2v = (col < HH2) ? b2g[col] : 0.0f;
        float wfv = WfS[col];                 // zero for col >= 40
        int mbase = mt * 16 + ((lane >> 4) << 3);
#pragma unroll
        for (int r = 0; r < 8; ++r) {
            float g = sigm(c[r] + b2v) * wfv;
            // reduce the 16 columns held by this half-wave group
            g += __shfl_xor(g, 1, 32);
            g += __shfl_xor(g, 2, 32);
            g += __shfl_xor(g, 4, 32);
            g += __shfl_xor(g, 8, 32);
            if ((lane & 15) == 0) atomicAdd(&lgS[mbase + r], g);
        }
    }
    __syncthreads();

    // ---------------- masked softmax over T (wave 0) -------------------------
    if (tid < 32) {
        float lv[7];
        float mx = -3.4e38f;
#pragma unroll
        for (int i = 0; i < 7; ++i) {
            int t = tid + 32 * i;
            float l = -3.4e38f;
            if (t < TT) {
                l = lgS[t];
                if (maskg[(size_t)b * TT + t] == 0) l = NEG_BIG;
            }
            lv[i] = l;
            mx = fmaxf(mx, l);
        }
        mx = fmaxf(mx, __shfl_xor(mx, 1, 32));
        mx = fmaxf(mx, __shfl_xor(mx, 2, 32));
        mx = fmaxf(mx, __shfl_xor(mx, 4, 32));
        mx = fmaxf(mx, __shfl_xor(mx, 8, 32));
        mx = fmaxf(mx, __shfl_xor(mx, 16, 32));
        float s = 0.0f;
#pragma unroll
        for (int i = 0; i < 7; ++i) {
            int t = tid + 32 * i;
            if (t < TT) {
                float e = __expf(lv[i] - mx);
                lv[i] = e;
                s += e;
            }
        }
        s += __shfl_xor(s, 1, 32);
        s += __shfl_xor(s, 2, 32);
        s += __shfl_xor(s, 4, 32);
        s += __shfl_xor(s, 8, 32);
        s += __shfl_xor(s, 16, 32);
        float inv = __builtin_amdgcn_rcpf(s);
#pragma unroll
        for (int i = 0; i < 7; ++i) {
            int t = tid + 32 * i;
            if (t < TT) outg[(size_t)b * TT + t] = lv[i] * inv;
        }
    }
}

extern "C" void kernel_launch(void* const* d_in, const int* in_sizes, int n_in,
                              void* d_out, int out_size, void* d_ws, size_t ws_size,
                              hipStream_t stream) {
    const float* qg     = (const float*)d_in[0];
    const float* kg     = (const float*)d_in[1];
    // d_in[2] = v  : unused by the reference output, deliberately not read
    const int*   maskg  = (const int*)d_in[3];
    const float* Wqg    = (const float*)d_in[4];
    const float* bqg    = (const float*)d_in[5];
    const float* alphag = (const float*)d_in[6];
    const float* W1g    = (const float*)d_in[7];
    const float* b1g    = (const float*)d_in[8];
    const float* W2g    = (const float*)d_in[9];
    const float* b2g    = (const float*)d_in[10];
    const float* Wfg    = (const float*)d_in[11];
    const float* bfg    = (const float*)d_in[12];
    float* outg = (float*)d_out;

    int B = in_sizes[0] / DD;   // 4096
    din_attention_kernel<<<B, 256, SMEM_BYTES, stream>>>(
        qg, kg, maskg, Wqg, bqg, alphag, W1g, b1g, W2g, b2g, Wfg, bfg, outg);
}
